// AttentionLSTM_74749610819711
// MI455X (gfx1250) — matrix operand order repaired and checked by reading, not compiled, not run on hardware
//
#include <hip/hip_runtime.h>

// CDNA5 (gfx1250) wave32 WMMA types
typedef __attribute__((ext_vector_type(16))) __bf16 v16bf;
typedef __attribute__((ext_vector_type(8)))  float  v8f;

constexpr int N_  = 128;
constexpr int T_  = 64;
constexpr int D_  = 1024;
constexpr int H_  = 1024;
constexpr int L_  = 16;     // DA*DA
constexpr int H4_ = 4096;   // 4*H

// fp32 -> bf16 round-to-nearest-even
__device__ __forceinline__ unsigned short f2bf(float x) {
  unsigned int u = __float_as_uint(x);
  u += 0x7FFFu + ((u >> 16) & 1u);
  return (unsigned short)(u >> 16);
}

// ---------------------------------------------------------------------------
// Fragment-swizzled index for an A-operand matrix (R x 1024, row tiles of 16,
// K tiles of 32). Matches the ISA 16-bit A-matrix lane layout:
//   lanes 0-15  : rows M=0..15, K = {0..7, 16..23}
//   lanes 16-31 : rows M=0..15, K = {8..15, 24..31}
// Each lane's 16 bf16 live contiguously (32B) -> fragment = 2x b128 loads.
// ---------------------------------------------------------------------------
__device__ __forceinline__ size_t swzA(int row, int k) {
  const int rt = row >> 4, kt = k >> 5, sub = k & 31;
  const int hi = (sub >> 3) & 1;
  const int lane = (row & 15) + hi * 16;
  const int jj = (sub & 7) + ((sub >= 16) ? 8 : 0);
  return ((size_t)(rt * 32 + kt) * 32 + lane) * 16 + jj;
}

// ---------------------------------------------------------------------------
// Weight swizzle+convert: W (K=1024 x 4096 fp32, row-major) -> bf16 B-fragment
// order. B layout: lanes 0-15 = cols N=0..15 holding K={0..7,16..23}; lanes 16-31
// hold K={8..15,24..31}. Tile (kt, ct) stored as 32 lanes x 16 bf16 contiguous (1 KB).
// One thread per (kt, ct, lane): 32*256*32 = 262144 threads.
// ---------------------------------------------------------------------------
__global__ __launch_bounds__(256)
void swz_w_kernel(const float* __restrict__ W, unsigned short* __restrict__ out) {
  const int idx = blockIdx.x * 256 + threadIdx.x;
  const int lane = idx & 31;
  const int ct = (idx >> 5) & 255;
  const int kt = idx >> 13;
  const int col = ct * 16 + (lane & 15);
  const int koff = (lane >> 4) * 8;
  unsigned short tmp[16];
  #pragma unroll
  for (int j = 0; j < 16; ++j)
    tmp[j] = f2bf(W[(size_t)(kt * 32 + koff + (j & 7) + (j >> 3) * 16) * H4_ + col]);
  uint4* dst = (uint4*)(out + ((size_t)(kt * 256 + ct) * 32 + lane) * 16);
  dst[0] = ((const uint4*)tmp)[0];
  dst[1] = ((const uint4*)tmp)[1];
}

// ---------------------------------------------------------------------------
// x swizzle+convert: x (N,T,D) fp32 -> per-timestep A-fragment order.
// One thread per (t, rt, kt, lane): 64*8*32*32 = 524288 threads.
// ---------------------------------------------------------------------------
__global__ __launch_bounds__(256)
void swz_x_kernel(const float* __restrict__ x, unsigned short* __restrict__ out) {
  const int idx = blockIdx.x * 256 + threadIdx.x;
  const int lane = idx & 31;
  const int kt = (idx >> 5) & 31;
  const int rt = (idx >> 10) & 7;
  const int t  = idx >> 13;
  const int row = rt * 16 + (lane & 15);
  const int koff = (lane >> 4) * 8;
  const float* src = x + ((size_t)row * T_ + t) * D_ + kt * 32 + koff;
  unsigned short tmp[16];
  #pragma unroll
  for (int j = 0; j < 8; ++j) { tmp[j] = f2bf(src[j]); tmp[8 + j] = f2bf(src[16 + j]); }
  uint4* dst = (uint4*)(out + (size_t)t * (N_ * D_) +
                        ((size_t)(rt * 32 + kt) * 32 + lane) * 16);
  dst[0] = ((const uint4*)tmp)[0];
  dst[1] = ((const uint4*)tmp)[1];
}

// ---------------------------------------------------------------------------
// h0 = c0 = mean of A over the 16 spatial cells; bf16 h0 written swizzled
// ---------------------------------------------------------------------------
__global__ __launch_bounds__(256)
void init_kernel(const float* __restrict__ A, float* __restrict__ h, float* __restrict__ c,
                 unsigned short* __restrict__ hb) {
  const int idx = blockIdx.x * 256 + threadIdx.x;  // over N*H
  const int n = idx >> 10, j = idx & 1023;
  const float* Ar = A + (size_t)idx * L_;
  float s = 0.f;
  #pragma unroll
  for (int l = 0; l < L_; ++l) s += Ar[l];
  s *= (1.f / 16.f);
  h[idx] = s;
  c[idx] = s;
  hb[swzA(n, j)] = f2bf(s);
}

// ---------------------------------------------------------------------------
// Per-step attention; emits bf16 attn in A-fragment swizzled order.
// ---------------------------------------------------------------------------
__global__ __launch_bounds__(256)
void attn_kernel(const float* __restrict__ A, const float* __restrict__ h,
                 unsigned short* __restrict__ attn_bf) {
  const int n = blockIdx.x, tid = threadIdx.x;
  __shared__ float red[L_];
  __shared__ float w[L_];
  const float* An = A + (size_t)n * H_ * L_;
  const float* hn = h + (size_t)n * H_;

  float part[L_];
  #pragma unroll
  for (int l = 0; l < L_; ++l) part[l] = 0.f;
  for (int hh = tid; hh < H_; hh += 256) {
    const float hv = hn[hh];
    const float* Ar = An + hh * L_;
    #pragma unroll
    for (int l = 0; l < L_; ++l) part[l] += hv * Ar[l];
  }
  if (tid < L_) red[tid] = 0.f;
  __syncthreads();
  #pragma unroll
  for (int l = 0; l < L_; ++l) atomicAdd(&red[l], part[l]);  // ds_add_f32
  __syncthreads();
  if (tid == 0) {
    const float scale = 0.03125f;  // 1/sqrt(1024)
    float sc[L_];
    float mx = -3.4e38f;
    #pragma unroll
    for (int l = 0; l < L_; ++l) { sc[l] = red[l] * scale; mx = fmaxf(mx, sc[l]); }
    float s = 0.f;
    #pragma unroll
    for (int l = 0; l < L_; ++l) { sc[l] = __expf(sc[l] - mx); s += sc[l]; }
    const float inv = 1.f / s;
    #pragma unroll
    for (int l = 0; l < L_; ++l) w[l] = sc[l] * inv;
  }
  __syncthreads();
  for (int hh = tid; hh < H_; hh += 256) {
    const float* Ar = An + hh * L_;
    float s = 0.f;
    #pragma unroll
    for (int l = 0; l < L_; ++l) s += Ar[l] * w[l];
    attn_bf[swzA(n, hh)] = f2bf(s);
  }
}

// ---------------------------------------------------------------------------
// Fused pre-activation GEMM (bf16 WMMA, fp32 accumulate):
//   a(N,4H) = x_t@Wx + h@Wh + attn@Wattn + b
// Block = 256 threads = 8 waves; each wave owns one 16x16 tile and streams its
// private 1 KB B-tiles through LDS via global_load_async_to_lds_b128 with
// double buffering (ASYNCcnt). No block barriers in the inner loop.
// Grid: (4096/128=32 col groups, 128/16=8 row tiles).
// ---------------------------------------------------------------------------
__global__ __launch_bounds__(256)
void gemm_gates_kernel(const unsigned short* __restrict__ xswz,
                       const unsigned short* __restrict__ hswz,
                       const unsigned short* __restrict__ aswz,
                       const unsigned short* __restrict__ Wxs,
                       const unsigned short* __restrict__ Whs,
                       const unsigned short* __restrict__ Was,
                       const float* __restrict__ bias,
                       float* __restrict__ aout, int t) {
  __shared__ __align__(16) unsigned char ldsB[8 * 2 * 1024];  // 8 waves x 2 bufs x 1KB
  const int tid = threadIdx.x;
  const int wave = tid >> 5;
  const int lane = tid & 31;
  const int rt = blockIdx.y;                 // row tile 0..7
  const int ct = blockIdx.x * 8 + wave;      // col tile 0..255

  unsigned char* myLds = ldsB + wave * 2048 + lane * 32;
  const unsigned ldsAddr = (unsigned)(uintptr_t)myLds;  // flat addr low 32 = LDS offset

  const unsigned short* A0 = xswz + (size_t)t * (N_ * D_);
  const unsigned short* A1 = hswz;
  const unsigned short* A2 = aswz;

  // Issue async copy of one wave-private B fragment tile (32B per lane).
  auto issueB = [&](int s) {
    const int m = s >> 5, kt = s & 31;
    const unsigned short* W = (m == 0) ? Wxs : (m == 1) ? Whs : Was;
    const unsigned short* g = W + ((size_t)(kt * 256 + ct) * 32 + lane) * 16;
    const unsigned l = ldsAddr + (unsigned)(s & 1) * 1024u;
    asm volatile("global_load_async_to_lds_b128 %0, %1, off\n\t"
                 "global_load_async_to_lds_b128 %0, %1, off offset:16"
                 :: "v"(l), "v"(g) : "memory");
  };

  v8f acc = {};
  issueB(0);

  #pragma unroll
  for (int s = 0; s < 96; ++s) {
    if (s + 1 < 96) {
      issueB(s + 1);                                          // prefetch next tile
      asm volatile("s_wait_asynccnt 0x2" ::: "memory");       // current tile landed
    } else {
      asm volatile("s_wait_asynccnt 0x0" ::: "memory");
    }
    const int m = s >> 5, kt = s & 31;
    const unsigned short* As = (m == 0) ? A0 : (m == 1) ? A1 : A2;

    union { v16bf v; uint4 q[2]; } fa;  // A fragment: 2x global b128
    const uint4* ga = (const uint4*)(As + ((size_t)(rt * 32 + kt) * 32 + lane) * 16);
    fa.q[0] = ga[0];
    fa.q[1] = ga[1];

    union { v16bf v; uint4 q[2]; } fb;  // B fragment: 2x ds b128
    const uint4* lb = (const uint4*)(ldsB + wave * 2048 + (s & 1) * 1024 + lane * 32);
    fb.q[0] = lb[0];
    fb.q[1] = lb[1];

    acc = __builtin_amdgcn_wmma_f32_16x16x32_bf16(
        /*neg_a=*/false, fa.v, /*neg_b=*/false, fb.v,
        /*c_mod=*/(short)0, acc, /*reuse_a=*/false, /*reuse_b=*/false);
  }

  // C/D layout: VGPR r, lanes 0-15 -> M=r; lanes 16-31 -> M=r+8.
  const int col = ct * 16 + (lane & 15);
  const float bc = bias[col];
  const int rb = rt * 16 + (lane >> 4) * 8;
  #pragma unroll
  for (int rr = 0; rr < 8; ++rr)
    aout[(size_t)(rb + rr) * H4_ + col] = acc[rr] + bc;
}

// ---------------------------------------------------------------------------
// LSTM pointwise: gates, cell/hidden update; h -> output + swizzled bf16 h
// ---------------------------------------------------------------------------
__global__ __launch_bounds__(256)
void lstm_kernel(const float* __restrict__ aacc, float* __restrict__ c,
                 float* __restrict__ h, unsigned short* __restrict__ hb,
                 float* __restrict__ out, int t) {
  const int idx = blockIdx.x * 256 + threadIdx.x;  // over N*H
  const int n = idx >> 10, j = idx & 1023;
  const float* an = aacc + (size_t)n * H4_;
  const float ig = 1.f / (1.f + __expf(-an[j]));
  const float fg = 1.f / (1.f + __expf(-an[H_ + j]));
  const float og = 1.f / (1.f + __expf(-an[2 * H_ + j]));
  const float gg = tanhf(an[3 * H_ + j]);
  const float cn = fg * c[idx] + ig * gg;
  const float hn = og * tanhf(cn);
  c[idx] = cn;
  h[idx] = hn;
  hb[swzA(n, j)] = f2bf(hn);
  out[((size_t)n * T_ + t) * H_ + j] = hn;   // (N, T, H)
}

// ---------------------------------------------------------------------------
// Host-side orchestration
// ---------------------------------------------------------------------------
extern "C" void kernel_launch(void* const* d_in, const int* in_sizes, int n_in,
                              void* d_out, int out_size, void* d_ws, size_t ws_size,
                              hipStream_t stream) {
  (void)in_sizes; (void)n_in; (void)out_size; (void)ws_size;
  const float* x     = (const float*)d_in[0];  // (N,T,D)
  const float* A     = (const float*)d_in[1];  // (N,H,4,4)
  const float* Wx    = (const float*)d_in[2];  // (D,4H)
  const float* Wh    = (const float*)d_in[3];  // (H,4H)
  const float* Wattn = (const float*)d_in[4];  // (H,4H)
  const float* b     = (const float*)d_in[5];  // (4H,)
  float* out = (float*)d_out;                  // (N,T,H)

  // Workspace layout (~43 MB)
  unsigned short* xswz = (unsigned short*)d_ws;                // N*T*D  bf16 (swizzled per t)
  unsigned short* Wxs  = xswz + (size_t)N_ * T_ * D_;          // D*4H   bf16 (B-swizzled)
  unsigned short* Whs  = Wxs + (size_t)D_ * H4_;
  unsigned short* Was  = Whs + (size_t)H_ * H4_;
  unsigned short* hb   = Was + (size_t)H_ * H4_;               // N*H    bf16 (A-swizzled)
  unsigned short* atb  = hb + (size_t)N_ * H_;                 // N*H    bf16 (A-swizzled)
  float* h    = (float*)(atb + (size_t)N_ * H_);               // N*H    f32
  float* c    = h + (size_t)N_ * H_;                           // N*H    f32
  float* aacc = c + (size_t)N_ * H_;                           // N*4H   f32

  swz_x_kernel<<<(T_ * 8 * 32 * 32) / 256, 256, 0, stream>>>(x, xswz);
  swz_w_kernel<<<(32 * 256 * 32) / 256, 256, 0, stream>>>(Wx, Wxs);
  swz_w_kernel<<<(32 * 256 * 32) / 256, 256, 0, stream>>>(Wh, Whs);
  swz_w_kernel<<<(32 * 256 * 32) / 256, 256, 0, stream>>>(Wattn, Was);

  init_kernel<<<(N_ * H_) / 256, 256, 0, stream>>>(A, h, c, hb);

  for (int t = 0; t < T_; ++t) {
    attn_kernel<<<N_, 256, 0, stream>>>(A, h, atb);
    gemm_gates_kernel<<<dim3(H4_ / 128, N_ / 16), 256, 0, stream>>>(
        xswz, hb, atb, Wxs, Whs, Was, b, aacc, t);
    lstm_kernel<<<(N_ * H_) / 256, 256, 0, stream>>>(aacc, c, h, hb, out, t);
  }
}
